// BigramLanguageModel_11261404250948
// MI455X (gfx1250) — compile-verified, matrix operand order checked by reading, not run on hardware
//
#include <hip/hip_runtime.h>
#include <hip/hip_bf16.h>
#include <math.h>

#define TB      1024          // sequence length
#define NBATCH  4
#define DIM     1024
#define NH      16
#define HDIM    64
#define NLAYER  8
#define FFDIM   (4 * DIM)
#define VOCAB   32000
#define ROWS    (NBATCH * TB) // 4096

typedef __attribute__((ext_vector_type(16))) __bf16 bf16x16;
typedef __attribute__((ext_vector_type(8)))  float  f32x8;

union BFrag { unsigned u[8]; bf16x16 v; };

__device__ __forceinline__ unsigned short f2bf(float f) {
  unsigned u = __float_as_uint(f);
  u += 0x7FFFu + ((u >> 16) & 1u);       // round-to-nearest-even
  return (unsigned short)(u >> 16);
}

// ISA 7.12.2: 16-bit A/B fragment k-pair base for VGPR v and lane-half.
__device__ __forceinline__ int kpair_of(int v, int half) {
  return ((v < 4) ? (v << 1) : (16 + ((v - 4) << 1))) + (half << 3);
}

// ---------------------------------------------------------------------------
// Tiled BF16 WMMA GEMM: C[M,N] = A[M,K](bf16) * B[K,N] (+bias)(+residual)(relu)
// Block = 256 threads (8 waves), tile 128x128, K-step 32.
// Register-staged double-buffered LDS: one barrier per K-step; next tile's
// global loads are in flight while the current tile's WMMAs execute.
// ---------------------------------------------------------------------------
template <bool B_BF16>
__global__ __launch_bounds__(256)
void gemm_bf16_wmma(const unsigned short* __restrict__ A,
                    const void* __restrict__ Bp,
                    const float* __restrict__ bias,
                    const float* __restrict__ residual,
                    float* __restrict__ outF,
                    unsigned short* __restrict__ outH,
                    int M, int N, int K, int do_relu) {
  __shared__ unsigned short lsA[2][128 * 40];   // [buf][row][k], padded rows
  __shared__ unsigned short lsB[2][128 * 40];   // [buf][col][k] (B transposed)

  const int tid  = threadIdx.x;
  const int lane = tid & 31, wave = tid >> 5;
  const int wm = wave & 1, wn = wave >> 1;      // 2x4 wave grid
  const int lc = lane & 15, lh = lane >> 4;
  const int rowBase = blockIdx.y * 128;
  const int colBase = blockIdx.x * 128;

  f32x8 acc[4][2];
#pragma unroll
  for (int i = 0; i < 4; ++i)
#pragma unroll
    for (int j = 0; j < 2; ++j)
      acc[i][j] = (f32x8){0.f, 0.f, 0.f, 0.f, 0.f, 0.f, 0.f, 0.f};

  uint4  regA[2];
  uint4  regBh[2];
  float4 regBf[4];

  auto gload = [&](int kk) {
#pragma unroll
    for (int it = 0; it < 2; ++it) {
      int c = tid + it * 256;
      int r = c >> 2, cc = (c & 3) << 3;
      regA[it] = *reinterpret_cast<const uint4*>(A + (size_t)(rowBase + r) * K + kk + cc);
    }
    if (B_BF16) {
      const unsigned short* Bh = (const unsigned short*)Bp;
#pragma unroll
      for (int it = 0; it < 2; ++it) {
        int c = tid + it * 256;
        int r = c >> 4, cc = (c & 15) << 3;
        regBh[it] = *reinterpret_cast<const uint4*>(Bh + (size_t)(kk + r) * N + colBase + cc);
      }
    } else {
      const float* Bf = (const float*)Bp;
#pragma unroll
      for (int it = 0; it < 4; ++it) {
        int c = tid + it * 256;
        int r = c >> 5, cc = (c & 31) << 2;
        regBf[it] = *reinterpret_cast<const float4*>(Bf + (size_t)(kk + r) * N + colBase + cc);
      }
    }
  };

  auto lstore = [&](int buf) {
#pragma unroll
    for (int it = 0; it < 2; ++it) {
      int c = tid + it * 256;
      int r = c >> 2, cc = (c & 3) << 3;
      *reinterpret_cast<uint4*>(&lsA[buf][r * 40 + cc]) = regA[it];
    }
    if (B_BF16) {
#pragma unroll
      for (int it = 0; it < 2; ++it) {
        int c = tid + it * 256;
        int r = c >> 4, cc = (c & 15) << 3;
        const unsigned short* e = reinterpret_cast<const unsigned short*>(&regBh[it]);
#pragma unroll
        for (int j = 0; j < 8; ++j) lsB[buf][(cc + j) * 40 + r] = e[j];
      }
    } else {
#pragma unroll
      for (int it = 0; it < 4; ++it) {
        int c = tid + it * 256;
        int r = c >> 5, cc = (c & 31) << 2;
        lsB[buf][(cc + 0) * 40 + r] = f2bf(regBf[it].x);
        lsB[buf][(cc + 1) * 40 + r] = f2bf(regBf[it].y);
        lsB[buf][(cc + 2) * 40 + r] = f2bf(regBf[it].z);
        lsB[buf][(cc + 3) * 40 + r] = f2bf(regBf[it].w);
      }
    }
  };

  gload(0);
  for (int kk = 0; kk < K; kk += 32) {
    const int buf = (kk >> 5) & 1;
    lstore(buf);
    __syncthreads();
    if (kk + 32 < K) gload(kk + 32);            // overlap with WMMAs below
    if (kk + 64 < K)                            // global_prefetch_b8 one ahead
      __builtin_prefetch(A + (size_t)(rowBase + (tid >> 2)) * K + kk + 64, 0, 1);

    BFrag bfr[2];
#pragma unroll
    for (int ni = 0; ni < 2; ++ni)
#pragma unroll
      for (int v = 0; v < 8; ++v)
        bfr[ni].u[v] = *reinterpret_cast<const unsigned*>(
            &lsB[buf][(wn * 32 + ni * 16 + lc) * 40 + kpair_of(v, lh)]);

#pragma unroll
    for (int mi = 0; mi < 4; ++mi) {
      BFrag afr;
#pragma unroll
      for (int v = 0; v < 8; ++v)
        afr.u[v] = *reinterpret_cast<const unsigned*>(
            &lsA[buf][(wm * 64 + mi * 16 + lc) * 40 + kpair_of(v, lh)]);
      acc[mi][0] = __builtin_amdgcn_wmma_f32_16x16x32_bf16(
          false, afr.v, false, bfr[0].v, (short)0, acc[mi][0], false, false);
      acc[mi][1] = __builtin_amdgcn_wmma_f32_16x16x32_bf16(
          false, afr.v, false, bfr[1].v, (short)0, acc[mi][1], false, false);
    }
    // single barrier per step: separates this step's frag reads of `buf`
    // from its overwrite at step kk+64, and orders lstore(buf^1) @ kk+32.
  }

  // Epilogue: C layout -> lane holds col lc, rows r + 8*lh
#pragma unroll
  for (int mi = 0; mi < 4; ++mi) {
#pragma unroll
    for (int ni = 0; ni < 2; ++ni) {
      int col = colBase + wn * 32 + ni * 16 + lc;
      float bv = bias ? bias[col] : 0.f;
#pragma unroll
      for (int r = 0; r < 8; ++r) {
        int row = rowBase + wm * 64 + mi * 16 + r + (lh << 3);
        float val = acc[mi][ni][r] + bv;
        if (residual) val += residual[(size_t)row * N + col];
        if (do_relu) val = fmaxf(val, 0.f);
        if (outF) outF[(size_t)row * N + col] = val;
        if (outH) outH[(size_t)row * N + col] = f2bf(val);
      }
    }
  }
}

// ---------------------------------------------------------------------------
// Flash attention: one wave per (b, h, 16-query tile); 32 keys per iteration.
// S = Q*K^T via 4 WMMAs, online softmax, O += P*V via 4 full-K32 WMMAs.
// scale = D^-0.5 = 1/32 (reference scales by full embedding dim).
// ---------------------------------------------------------------------------
__global__ __launch_bounds__(32)
void attn_kernel(const unsigned short* __restrict__ Q,
                 const unsigned short* __restrict__ Kb,
                 const unsigned short* __restrict__ V,
                 unsigned short* __restrict__ O) {
  __shared__ unsigned short p_lds[16 * 32];     // P tile (16 q x 32 keys)
  const int lane = threadIdx.x;
  const int lc = lane & 15, lh = lane >> 4;
  const int gid = blockIdx.x;
  const int qt = gid & 63;
  const int h  = (gid >> 6) & 15;
  const int b  = gid >> 10;
  const int bbase   = b * TB;
  const int rowbase = bbase + qt * 16;
  const int hbase   = h * HDIM;

  BFrag qf[2];
#pragma unroll
  for (int kh = 0; kh < 2; ++kh)
#pragma unroll
    for (int v = 0; v < 8; ++v)
      qf[kh].u[v] = *reinterpret_cast<const unsigned*>(
          Q + (size_t)(rowbase + lc) * DIM + hbase + kh * 32 + kpair_of(v, lh));

  f32x8 oacc[4];
#pragma unroll
  for (int i = 0; i < 4; ++i) oacc[i] = (f32x8){0.f, 0.f, 0.f, 0.f, 0.f, 0.f, 0.f, 0.f};
  float mrow[8], lrow[8];
#pragma unroll
  for (int r = 0; r < 8; ++r) { mrow[r] = -__builtin_inff(); lrow[r] = 0.f; }

  const int nblk = (qt + 2) >> 1;               // ceil((qt+1)/2) 32-key blocks
  for (int jb = 0; jb < nblk; ++jb) {
    const int kbase0 = jb * 32;

    f32x8 s[2];
#pragma unroll
    for (int t16 = 0; t16 < 2; ++t16) {
      BFrag kf0, kf1;
#pragma unroll
      for (int v = 0; v < 8; ++v) {
        kf0.u[v] = *reinterpret_cast<const unsigned*>(
            Kb + (size_t)(bbase + kbase0 + t16 * 16 + lc) * DIM + hbase + kpair_of(v, lh));
        kf1.u[v] = *reinterpret_cast<const unsigned*>(
            Kb + (size_t)(bbase + kbase0 + t16 * 16 + lc) * DIM + hbase + 32 + kpair_of(v, lh));
      }
      s[t16] = (f32x8){0.f, 0.f, 0.f, 0.f, 0.f, 0.f, 0.f, 0.f};
      s[t16] = __builtin_amdgcn_wmma_f32_16x16x32_bf16(false, qf[0].v, false, kf0.v, (short)0, s[t16], false, false);
      s[t16] = __builtin_amdgcn_wmma_f32_16x16x32_bf16(false, qf[1].v, false, kf1.v, (short)0, s[t16], false, false);
    }

#pragma unroll
    for (int r = 0; r < 8; ++r) {
      int m = r + (lh << 3);
      int qidx = qt * 16 + m;
      float sv0 = s[0][r] * 0.03125f;
      float sv1 = s[1][r] * 0.03125f;
      if (kbase0 + lc      > qidx) sv0 = -__builtin_inff();   // causal mask
      if (kbase0 + 16 + lc > qidx) sv1 = -__builtin_inff();
      float t = fmaxf(sv0, sv1);
      t = fmaxf(t, __shfl_xor(t, 1, 32));
      t = fmaxf(t, __shfl_xor(t, 2, 32));
      t = fmaxf(t, __shfl_xor(t, 4, 32));
      t = fmaxf(t, __shfl_xor(t, 8, 32));
      float mnew  = fmaxf(mrow[r], t);
      float alpha = __expf(mrow[r] - mnew);
      mrow[r] = mnew;
      float p0 = __expf(sv0 - mnew);
      float p1 = __expf(sv1 - mnew);
      float ps = p0 + p1;
      ps += __shfl_xor(ps, 1, 32);
      ps += __shfl_xor(ps, 2, 32);
      ps += __shfl_xor(ps, 4, 32);
      ps += __shfl_xor(ps, 8, 32);
      lrow[r] = lrow[r] * alpha + ps;
#pragma unroll
      for (int nt = 0; nt < 4; ++nt) oacc[nt][r] *= alpha;
      p_lds[m * 32 + lc]      = f2bf(p0);
      p_lds[m * 32 + 16 + lc] = f2bf(p1);
    }
    __syncthreads();

    BFrag pf;   // C-layout -> A-layout reshape via LDS (full 16x32, no padding)
#pragma unroll
    for (int v = 0; v < 8; ++v)
      pf.u[v] = *reinterpret_cast<const unsigned*>(&p_lds[lc * 32 + kpair_of(v, lh)]);

#pragma unroll
    for (int nt = 0; nt < 4; ++nt) {
      BFrag vf;   // B fragment: 32 real keys
#pragma unroll
      for (int v = 0; v < 8; ++v) {
        int kp = kpair_of(v, lh);
        unsigned short a0 = V[(size_t)(bbase + kbase0 + kp)     * DIM + hbase + nt * 16 + lc];
        unsigned short a1 = V[(size_t)(bbase + kbase0 + kp + 1) * DIM + hbase + nt * 16 + lc];
        vf.u[v] = (unsigned)a0 | ((unsigned)a1 << 16);
      }
      oacc[nt] = __builtin_amdgcn_wmma_f32_16x16x32_bf16(false, pf.v, false, vf.v, (short)0, oacc[nt], false, false);
    }
    __syncthreads();
  }

#pragma unroll
  for (int nt = 0; nt < 4; ++nt)
#pragma unroll
    for (int r = 0; r < 8; ++r) {
      int m = r + (lh << 3);
      float val = oacc[nt][r] / lrow[r];
      O[(size_t)(rowbase + m) * DIM + hbase + nt * 16 + lc] = f2bf(val);
    }
}

// ---------------------------------------------------------------------------
// LayerNorm over D=1024 (one block per row), writes bf16 for the next GEMM A.
// ---------------------------------------------------------------------------
__global__ __launch_bounds__(256)
void layernorm_kernel(const float* __restrict__ x, const float* __restrict__ g,
                      const float* __restrict__ bta, unsigned short* __restrict__ out) {
  __shared__ float red[256];
  const int row = blockIdx.x, tid = threadIdx.x;
  const float* xr = x + (size_t)row * DIM;
  const int i = tid * 4;
  float4 v = *reinterpret_cast<const float4*>(xr + i);
  red[tid] = v.x + v.y + v.z + v.w;
  __syncthreads();
  for (int o = 128; o > 0; o >>= 1) { if (tid < o) red[tid] += red[tid + o]; __syncthreads(); }
  float mu = red[0] * (1.f / DIM);
  __syncthreads();
  float dx = v.x - mu, dy = v.y - mu, dz = v.z - mu, dw = v.w - mu;
  red[tid] = dx * dx + dy * dy + dz * dz + dw * dw;
  __syncthreads();
  for (int o = 128; o > 0; o >>= 1) { if (tid < o) red[tid] += red[tid + o]; __syncthreads(); }
  float rstd = rsqrtf(red[0] * (1.f / DIM) + 1e-5f);
  float4 gv = *reinterpret_cast<const float4*>(g + i);
  float4 bv = *reinterpret_cast<const float4*>(bta + i);
  unsigned short* orow = out + (size_t)row * DIM;
  orow[i + 0] = f2bf(dx * rstd * gv.x + bv.x);
  orow[i + 1] = f2bf(dy * rstd * gv.y + bv.y);
  orow[i + 2] = f2bf(dz * rstd * gv.z + bv.z);
  orow[i + 3] = f2bf(dw * rstd * gv.w + bv.w);
}

__global__ __launch_bounds__(256)
void embed_kernel(const int* __restrict__ idx, const float* __restrict__ tok,
                  const float* __restrict__ pos, float* __restrict__ x) {
  int row = blockIdx.x;
  int t = row & (TB - 1);
  int token = idx[row];
  int i = threadIdx.x * 4;
  float4 a = *reinterpret_cast<const float4*>(tok + (size_t)token * DIM + i);
  float4 p = *reinterpret_cast<const float4*>(pos + (size_t)t * DIM + i);
  float4 o;
  o.x = a.x + p.x; o.y = a.y + p.y; o.z = a.z + p.z; o.w = a.w + p.w;
  *reinterpret_cast<float4*>(x + (size_t)row * DIM + i) = o;
}

// (L,H,D,HD) f32 -> per-layer [D][H*HD] bf16 (head-concat column order)
__global__ __launch_bounds__(256)
void relayout_qkv_kernel(const float* __restrict__ w, unsigned short* __restrict__ o) {
  size_t f = (size_t)blockIdx.x * 256 + threadIdx.x;    // over L*D*D = 8M
  int l = (int)(f >> 20);
  int rem = (int)(f & ((1u << 20) - 1));
  int k = rem >> 10;
  int n = rem & 1023;
  int h = n >> 6, e = n & 63;
  size_t in = (((size_t)l * NH + h) * DIM + k) * HDIM + e;
  o[f] = f2bf(w[in]);
}

__global__ __launch_bounds__(256)
void nll_kernel(const float* __restrict__ logits, const int* __restrict__ targets,
                float* __restrict__ nll) {
  __shared__ float red[256];
  const int row = blockIdx.x, tid = threadIdx.x;
  const float* lr = logits + (size_t)row * VOCAB;
  float mx = -__builtin_inff();
  for (int i = tid; i < VOCAB; i += 256) mx = fmaxf(mx, lr[i]);
  red[tid] = mx;
  __syncthreads();
  for (int o = 128; o > 0; o >>= 1) { if (tid < o) red[tid] = fmaxf(red[tid], red[tid + o]); __syncthreads(); }
  mx = red[0];
  __syncthreads();
  float s = 0.f;
  for (int i = tid; i < VOCAB; i += 256) s += __expf(lr[i] - mx);
  red[tid] = s;
  __syncthreads();
  for (int o = 128; o > 0; o >>= 1) { if (tid < o) red[tid] += red[tid + o]; __syncthreads(); }
  if (tid == 0) {
    int t = targets[row];
    nll[row] = -(lr[t] - mx - __logf(red[0]));
  }
}

__global__ __launch_bounds__(256)
void loss_reduce_kernel(const float* __restrict__ nll, float* __restrict__ out) {
  __shared__ float red[256];
  const int tid = threadIdx.x;
  float s = 0.f;
  for (int i = tid; i < ROWS; i += 256) s += nll[i];
  red[tid] = s;
  __syncthreads();
  for (int o = 128; o > 0; o >>= 1) { if (tid < o) red[tid] += red[tid + o]; __syncthreads(); }
  if (tid == 0) out[0] = red[0] * (1.f / ROWS);
}

// ---------------------------------------------------------------------------
extern "C" void kernel_launch(void* const* d_in, const int* in_sizes, int n_in,
                              void* d_out, int out_size, void* d_ws, size_t ws_size,
                              hipStream_t stream) {
  (void)in_sizes; (void)n_in; (void)out_size; (void)ws_size;
  const int*   idx     = (const int*)d_in[0];
  const int*   targets = (const int*)d_in[1];
  const float* tok_emb = (const float*)d_in[2];
  const float* pos_emb = (const float*)d_in[3];
  const float* wq    = (const float*)d_in[4];
  const float* wk    = (const float*)d_in[5];
  const float* wv    = (const float*)d_in[6];
  const float* wo    = (const float*)d_in[7];
  const float* bo    = (const float*)d_in[8];
  const float* w1    = (const float*)d_in[9];
  const float* b1    = (const float*)d_in[10];
  const float* w2    = (const float*)d_in[11];
  const float* b2    = (const float*)d_in[12];
  const float* ln1_g = (const float*)d_in[13];
  const float* ln1_b = (const float*)d_in[14];
  const float* ln2_g = (const float*)d_in[15];
  const float* ln2_b = (const float*)d_in[16];
  const float* lnf_g = (const float*)d_in[17];
  const float* lnf_b = (const float*)d_in[18];
  const float* w_lm  = (const float*)d_in[19];
  const float* b_lm  = (const float*)d_in[20];

  float* logits = (float*)d_out;
  float* loss   = (float*)d_out + (size_t)ROWS * VOCAB;

  char* ws = (char*)d_ws;
  size_t off = 0;
  auto take = [&](size_t bytes) -> char* {
    char* p = ws + off;
    off += (bytes + 255) & ~(size_t)255;
    return p;
  };

  float*          x    = (float*)take((size_t)ROWS * DIM * 4);
  unsigned short* hb   = (unsigned short*)take((size_t)ROWS * DIM * 2);
  unsigned short* qb   = (unsigned short*)take((size_t)ROWS * DIM * 2);
  unsigned short* kb   = (unsigned short*)take((size_t)ROWS * DIM * 2);
  unsigned short* vb   = (unsigned short*)take((size_t)ROWS * DIM * 2);
  unsigned short* ob   = (unsigned short*)take((size_t)ROWS * DIM * 2);
  unsigned short* ffb  = (unsigned short*)take((size_t)ROWS * FFDIM * 2);
  unsigned short* wqh  = (unsigned short*)take((size_t)NLAYER * DIM * DIM * 2);
  unsigned short* wkh  = (unsigned short*)take((size_t)NLAYER * DIM * DIM * 2);
  unsigned short* wvh  = (unsigned short*)take((size_t)NLAYER * DIM * DIM * 2);
  float*          nllb = (float*)take((size_t)ROWS * 4);

  const dim3 blk(256);
  const dim3 gRelayout((NLAYER * DIM * DIM) / 256);
  relayout_qkv_kernel<<<gRelayout, blk, 0, stream>>>(wq, wqh);
  relayout_qkv_kernel<<<gRelayout, blk, 0, stream>>>(wk, wkh);
  relayout_qkv_kernel<<<gRelayout, blk, 0, stream>>>(wv, wvh);
  embed_kernel<<<dim3(ROWS), blk, 0, stream>>>(idx, tok_emb, pos_emb, x);

  const dim3 gProj(DIM / 128, ROWS / 128);     // 8 x 32
  const dim3 gFF1(FFDIM / 128, ROWS / 128);    // 32 x 32
  const dim3 gLm(VOCAB / 128, ROWS / 128);     // 250 x 32
  const dim3 gAttn(NBATCH * NH * (TB / 16));   // 4096 waves

  for (int l = 0; l < NLAYER; ++l) {
    layernorm_kernel<<<dim3(ROWS), blk, 0, stream>>>(x, ln1_g + l * DIM, ln1_b + l * DIM, hb);
    gemm_bf16_wmma<true><<<gProj, blk, 0, stream>>>(hb, wqh + (size_t)l * DIM * DIM,
        nullptr, nullptr, nullptr, qb, ROWS, DIM, DIM, 0);
    gemm_bf16_wmma<true><<<gProj, blk, 0, stream>>>(hb, wkh + (size_t)l * DIM * DIM,
        nullptr, nullptr, nullptr, kb, ROWS, DIM, DIM, 0);
    gemm_bf16_wmma<true><<<gProj, blk, 0, stream>>>(hb, wvh + (size_t)l * DIM * DIM,
        nullptr, nullptr, nullptr, vb, ROWS, DIM, DIM, 0);
    attn_kernel<<<gAttn, dim3(32), 0, stream>>>(qb, kb, vb, ob);
    gemm_bf16_wmma<false><<<gProj, blk, 0, stream>>>(ob, wo + (size_t)l * DIM * DIM,
        bo + l * DIM, x, x, nullptr, ROWS, DIM, DIM, 0);
    layernorm_kernel<<<dim3(ROWS), blk, 0, stream>>>(x, ln2_g + l * DIM, ln2_b + l * DIM, hb);
    gemm_bf16_wmma<false><<<gFF1, blk, 0, stream>>>(hb, w1 + (size_t)l * DIM * FFDIM,
        b1 + (size_t)l * FFDIM, nullptr, nullptr, ffb, ROWS, FFDIM, DIM, 1);
    gemm_bf16_wmma<false><<<gProj, blk, 0, stream>>>(ffb, w2 + (size_t)l * FFDIM * DIM,
        b2 + l * DIM, x, x, nullptr, ROWS, DIM, FFDIM, 0);
  }

  layernorm_kernel<<<dim3(ROWS), blk, 0, stream>>>(x, lnf_g, lnf_b, hb);
  gemm_bf16_wmma<false><<<gLm, blk, 0, stream>>>(hb, w_lm, b_lm, nullptr,
      logits, nullptr, ROWS, VOCAB, DIM, 0);
  nll_kernel<<<dim3(ROWS), blk, 0, stream>>>(logits, targets, nllb);
  loss_reduce_kernel<<<dim3(1), blk, 0, stream>>>(nllb, loss);
}
